// WeightsLib2DAlpha_91242285236733
// MI455X (gfx1250) — matrix operand order repaired and checked by reading, not compiled, not run on hardware
//
#include <hip/hip_runtime.h>

#define LL   1024
#define NN   8192
#define BB   64
#define EPSF 1e-4f

typedef __attribute__((ext_vector_type(16))) __bf16 v16bf;
typedef __attribute__((ext_vector_type(8)))  float  v8f;

struct U128 { unsigned int a, b, c, d; };

union BfVec {
  v16bf v;
  __bf16 h[16];
  unsigned short u[16];
  U128 q[2];
};

static __device__ __forceinline__ unsigned short f2bfu(float f) {
  union { __bf16 h; unsigned short u; } c;
  c.h = (__bf16)f;                 // hardware cvt path, RNE
  return c.u;
}

// ---------------------------------------------------------------------------
// Kernel 0b: one-time x f32 -> bf16 conversion (64x1024).
// ---------------------------------------------------------------------------
__global__ __launch_bounds__(256)
void k_xcast(const float* __restrict__ x, unsigned short* __restrict__ xb) {
  size_t t = (size_t)blockIdx.x * 256 + threadIdx.x;   // 16384 threads x 4 elems
  float4 v = *(const float4*)(x + t * 4);
  unsigned p0 = (unsigned)f2bfu(v.x) | ((unsigned)f2bfu(v.y) << 16);
  unsigned p1 = (unsigned)f2bfu(v.z) | ((unsigned)f2bfu(v.w) << 16);
  unsigned long long pk = ((unsigned long long)p1 << 32) | p0;
  *(unsigned long long*)(xb + t * 4) = pk;
}

// ---------------------------------------------------------------------------
// Kernel 1: fused dual GEMM (xt, xg) + SiLU gate + |mod| group sums.
// Grid: 128 blocks (N tiles of 64), 256 threads = 8 waves.
// Wave w: N strip n16 = w&3, M half mh = w>>2 (2 of 4 M tiles) -> 1024 waves
// in flight for the HBM-bound W stream. x (bf16) staged into LDS with
// GLOBAL_LOAD_ASYNC_TO_LDS_B128 (ASYNCcnt).
// ---------------------------------------------------------------------------
__global__ __launch_bounds__(256)
void k_gemm_mod(const unsigned short* __restrict__ xb,
                const float* __restrict__ Wt,
                const float* __restrict__ Wg,
                float* __restrict__ mod,
                float* __restrict__ abs_sums)
{
  __shared__ unsigned short xlds[BB * 128];   // bf16 x chunk: 64 rows x 128 K
  __shared__ float lsum[BB];

  const int tid  = threadIdx.x;
  const int lane = tid & 31;
  const int wave = tid >> 5;
  const int lo   = lane & 15;
  const int hi   = lane >> 4;
  const int n16  = wave & 3;        // N strip within the block tile
  const int mh   = wave >> 2;       // M half: tiles {2mh, 2mh+1}
  const int nBase = blockIdx.x * 64;

  if (tid < BB) lsum[tid] = 0.0f;

  v8f acc_t[2] = {};
  v8f acc_g[2] = {};

  const int rowW = nBase + n16 * 16 + lo;           // B-fragment: lane -> N
  const float* wtp = Wt + (size_t)rowW * LL;
  const float* wgp = Wg + (size_t)rowW * LL;

  for (int kc = 0; kc < LL; kc += 128) {
    __syncthreads();   // all waves done reading the previous chunk

    // async-copy x_bf16[0:64, kc:kc+128] into LDS: 1024 x 16B segments
    #pragma unroll
    for (int f = 0; f < 4; ++f) {
      int s   = f * 256 + tid;
      int row = s >> 4;
      int seg = s & 15;
      const unsigned short* gp = xb + (size_t)row * LL + kc + seg * 8;
      unsigned ldsoff = (unsigned)(size_t)(const void*)&xlds[row * 128 + seg * 8];
      asm volatile("global_load_async_to_lds_b128 %0, %1, off"
                   :: "v"(ldsoff), "v"(gp) : "memory");
    }
    asm volatile("s_wait_asynccnt 0x0" ::: "memory");
    __syncthreads();

    // prefetch next W chunk (read-once stream)
    if (kc + 128 < LL) {
      __builtin_prefetch(wtp + kc + 128, 0, 0);
      __builtin_prefetch(wgp + kc + 128, 0, 0);
    }

    #pragma unroll
    for (int ks = 0; ks < 128; ks += 32) {
      // B fragments (32x16, lane = N, 16 contiguous K per lane per half)
      BfVec bt, bg;
      {
        const int col = kc + ks + hi * 16;
        const float4* pt = (const float4*)(wtp + col);
        const float4* pg = (const float4*)(wgp + col);
        #pragma unroll
        for (int q = 0; q < 4; ++q) {
          float4 a = pt[q], b = pg[q];
          bt.h[q*4+0] = (__bf16)a.x; bt.h[q*4+1] = (__bf16)a.y;
          bt.h[q*4+2] = (__bf16)a.z; bt.h[q*4+3] = (__bf16)a.w;
          bg.h[q*4+0] = (__bf16)b.x; bg.h[q*4+1] = (__bf16)b.y;
          bg.h[q*4+2] = (__bf16)b.z; bg.h[q*4+3] = (__bf16)b.w;
        }
      }
      // A fragments (16x32): lane holds row M=lo, 16 contiguous K per half
      #pragma unroll
      for (int t = 0; t < 2; ++t) {
        const int mt = mh * 2 + t;
        BfVec av;
        const int row = mt * 16 + lo;
        const int off = ks + hi * 16;
        av.q[0] = *(const U128*)&xlds[row * 128 + off];
        av.q[1] = *(const U128*)&xlds[row * 128 + off + 8];
        acc_t[t] = __builtin_amdgcn_wmma_f32_16x16x32_bf16(
            false, av.v, false, bt.v, (short)0, acc_t[t], false, false);
        acc_g[t] = __builtin_amdgcn_wmma_f32_16x16x32_bf16(
            false, av.v, false, bg.v, (short)0, acc_g[t], false, false);
      }
    }
  }

  // Epilogue: mod = xt * silu(xg); accumulate |mod| per (row, group)
  const int g   = nBase >> 11;                 // n / 2048
  const int col = nBase + n16 * 16 + lo;
  #pragma unroll
  for (int t = 0; t < 2; ++t) {
    const int mt = mh * 2 + t;
    #pragma unroll
    for (int r = 0; r < 8; ++r) {
      float vt  = acc_t[t][r];
      float vg  = acc_g[t][r];
      float sig = __builtin_amdgcn_rcpf(1.0f + __expf(-vg));
      float m   = vt * vg * sig;
      int row   = mt * 16 + hi * 8 + r;        // C layout: M = r + 8*hi
      mod[(size_t)row * NN + col] = m;
      float a = fabsf(m);
      a += __shfl_xor(a, 1, 32);
      a += __shfl_xor(a, 2, 32);
      a += __shfl_xor(a, 4, 32);
      a += __shfl_xor(a, 8, 32);
      if (lo == 0) atomicAdd(&lsum[row], a);
    }
  }
  __syncthreads();
  if (tid < BB) atomicAdd(&abs_sums[tid * 4 + g], lsum[tid]);
}

// ---------------------------------------------------------------------------
__global__ void k_zero(float* __restrict__ sums, float* __restrict__ magsum) {
  int t = threadIdx.x;
  sums[t] = 0.0f;
  if (t < BB) magsum[t] = 0.0f;
}

__global__ void k_scale(const float* __restrict__ sums, float* __restrict__ scale) {
  int t = threadIdx.x;   // 256 = 64 b x 4 k
  scale[t] = rsqrtf(sums[t] * (1.0f / 2048.0f) + EPSF);
}

// ---------------------------------------------------------------------------
// Kernel 3: per-b sum of sqrt(re^2+im^2+eps). Recomputes the rank-4
// outer-product contraction on the fly from the 2MB "mod" tensor.
// Grid: (64 b, 16 i-tiles), 256 threads = j.
// ---------------------------------------------------------------------------
__global__ __launch_bounds__(256)
void k_mag(const float* __restrict__ mod, const float* __restrict__ scale,
           const float* __restrict__ wstat, const float* __restrict__ wmod,
           const float* __restrict__ cw, const float* __restrict__ cb,
           float* __restrict__ magsum)
{
  __shared__ float aV[2 * 4 * 16];
  __shared__ float red[256];
  const int b  = blockIdx.x;
  const int ib = blockIdx.y * 16;
  const int j  = threadIdx.x;

  float we[4];
  #pragma unroll
  for (int r = 0; r < 4; ++r) we[r] = cw[r] + cw[4 + r] + cw[8 + r] + cw[12 + r];
  const float be = cb[0] + cb[1] + cb[2] + cb[3];

  if (j < 128) {
    int k = j >> 6, r = (j >> 4) & 3, ii = j & 15;
    aV[j] = mod[(size_t)b * NN + k * 2048 + r * 512 + ib + ii] * scale[b * 4 + k];
  }
  float bk[2][4];
  #pragma unroll
  for (int k = 0; k < 2; ++k)
    #pragma unroll
    for (int r = 0; r < 4; ++r)
      bk[k][r] = mod[(size_t)b * NN + k * 2048 + r * 512 + 256 + j] * scale[b * 4 + k];
  __syncthreads();

  float s = 0.0f;
  for (int ii = 0; ii < 16; ++ii) {
    int i = ib + ii;
    float m0 = 0.0f, m1 = 0.0f;
    #pragma unroll
    for (int r = 0; r < 4; ++r) {
      m0 += we[r] * aV[(0 * 4 + r) * 16 + ii] * bk[0][r];
      m1 += we[r] * aV[(1 * 4 + r) * 16 + ii] * bk[1][r];
    }
    m0 = (m0 + be) * wmod[(0 * 256 + i) * 256 + j];
    m1 = (m1 + be) * wmod[(1 * 256 + i) * 256 + j];
    float a1r = wstat[(((size_t)1 * 256 + i) * 256 + j) * 2 + 0];
    float a1i = wstat[(((size_t)1 * 256 + i) * 256 + j) * 2 + 1];
    float re = a1r * m0 - a1i * m1;
    float im = a1r * m1 - a1i * m0;
    s += sqrtf(re * re + im * im + EPSF);
  }
  red[j] = s;
  __syncthreads();
  for (int st = 128; st > 0; st >>= 1) {
    if (j < st) red[j] += red[j + st];
    __syncthreads();
  }
  if (j == 0) atomicAdd(&magsum[b], red[0]);
}

__global__ void k_inv(const float* __restrict__ magsum, float* __restrict__ invm) {
  int t = threadIdx.x;   // 64
  invm[t] = rsqrtf(magsum[t] * (1.0f / 65536.0f) + EPSF);
}

// ---------------------------------------------------------------------------
// Kernel 5: final output. Same tiling as k_mag, all 4 k-channels.
// ---------------------------------------------------------------------------
__global__ __launch_bounds__(256)
void k_out(const float* __restrict__ mod, const float* __restrict__ scale,
           const float* __restrict__ wstat, const float* __restrict__ wmod,
           const float* __restrict__ cw, const float* __restrict__ cb,
           const float* __restrict__ invm, float* __restrict__ out)
{
  __shared__ float aV[4 * 4 * 16];
  const int b  = blockIdx.x;
  const int ib = blockIdx.y * 16;
  const int j  = threadIdx.x;

  float we[4];
  #pragma unroll
  for (int r = 0; r < 4; ++r) we[r] = cw[r] + cw[4 + r] + cw[8 + r] + cw[12 + r];
  const float be = cb[0] + cb[1] + cb[2] + cb[3];

  {
    int k = j >> 6, r = (j >> 4) & 3, ii = j & 15;
    aV[j] = mod[(size_t)b * NN + k * 2048 + r * 512 + ib + ii] * scale[b * 4 + k];
  }
  float bk[4][4];
  #pragma unroll
  for (int k = 0; k < 4; ++k)
    #pragma unroll
    for (int r = 0; r < 4; ++r)
      bk[k][r] = mod[(size_t)b * NN + k * 2048 + r * 512 + 256 + j] * scale[b * 4 + k];
  __syncthreads();

  const float inv = invm[b];
  for (int ii = 0; ii < 16; ++ii) {
    int i = ib + ii;
    float mk[4];
    #pragma unroll
    for (int k = 0; k < 4; ++k) {
      float s = 0.0f;
      #pragma unroll
      for (int r = 0; r < 4; ++r)
        s += we[r] * aV[(k * 4 + r) * 16 + ii] * bk[k][r];
      mk[k] = (s + be) * wmod[((size_t)k * 256 + i) * 256 + j];
    }
    float a1r = wstat[(((size_t)1 * 256 + i) * 256 + j) * 2 + 0];
    float a1i = wstat[(((size_t)1 * 256 + i) * 256 + j) * 2 + 1];
    float re = a1r * mk[0] - a1i * mk[1];
    float im = a1r * mk[1] - a1i * mk[0];
    float a0r = wstat[(((size_t)0 * 256 + i) * 256 + j) * 2 + 0];
    float a0i = wstat[(((size_t)0 * 256 + i) * 256 + j) * 2 + 1];
    float rden = rsqrtf(mk[2] * mk[2] + mk[3] * mk[3] + EPSF);
    float ct = mk[2] * rden;
    float st = mk[3] * rden;
    out[(size_t)b * 65536 + (size_t)i * 256 + j] =
        (a0r + re * inv) * ct + (a0i + im * inv) * st;
  }
}

// ---------------------------------------------------------------------------
extern "C" void kernel_launch(void* const* d_in, const int* in_sizes, int n_in,
                              void* d_out, int out_size, void* d_ws, size_t ws_size,
                              hipStream_t stream) {
  const float* x   = (const float*)d_in[0];
  const float* Wt  = (const float*)d_in[1];
  const float* Wg  = (const float*)d_in[2];
  const float* wst = (const float*)d_in[3];
  const float* wm  = (const float*)d_in[4];
  const float* cw  = (const float*)d_in[5];
  const float* cb  = (const float*)d_in[6];
  float* out = (float*)d_out;

  float* ws     = (float*)d_ws;
  float* mod    = ws;              // 64*8192 = 524288 floats
  float* sums   = ws + 524288;     // 256
  float* scale  = ws + 524544;     // 256
  float* magsum = ws + 524800;     // 64
  float* invm   = ws + 524864;     // 64
  unsigned short* xb = (unsigned short*)(ws + 524928);  // 64*1024 bf16

  k_zero<<<1, 256, 0, stream>>>(sums, magsum);
  k_xcast<<<64, 256, 0, stream>>>(x, xb);
  k_gemm_mod<<<128, 256, 0, stream>>>(xb, Wt, Wg, mod, sums);
  k_scale<<<1, 256, 0, stream>>>(sums, scale);
  k_mag<<<dim3(64, 16), 256, 0, stream>>>(mod, scale, wst, wm, cw, cb, magsum);
  k_inv<<<1, 64, 0, stream>>>(magsum, invm);
  k_out<<<dim3(64, 16), 256, 0, stream>>>(mod, scale, wst, wm, cw, cb, invm, out);
}
